// GCN_with_feature_39281770889758
// MI455X (gfx1250) — matrix operand order repaired and checked by reading, not compiled, not run on hardware
//
#include <hip/hip_runtime.h>

// ---------------------------------------------------------------------------
// GCN forward for MI455X (gfx1250, wave32).
//  - dense transforms: v_wmma_f32_16x16x32_bf16, A staged in LDS via
//    global_load_async_to_lds_b128 (ASYNCcnt path)
//  - edge propagation: wave-per-edge f32 atomics (L2-resident working set)
//  - bias/residual fused into the layernorm pass (one sweep per step)
// ---------------------------------------------------------------------------

typedef __attribute__((ext_vector_type(16))) __bf16       v16bf;
typedef __attribute__((ext_vector_type(8)))  float        v8f;
typedef __attribute__((ext_vector_type(8)))  unsigned int v8u;

union ABu { v8u u; v16bf h; };

__device__ __forceinline__ unsigned short f32_to_bf16_bits(float f) {
    union { __bf16 b; unsigned short u; } cv;
    cv.b = (__bf16)f;               // RNE convert
    return cv.u;
}

// ---- f32 -> bf16 bulk convert ---------------------------------------------
__global__ void k_cvt_bf16(const float* __restrict__ in,
                           unsigned short* __restrict__ out, int n) {
    int i = blockIdx.x * 256 + threadIdx.x;
    if (i < n) out[i] = f32_to_bf16_bits(in[i]);
}

// ---- pack weight matrix W[K][NF] into WMMA B-fragment order ----------------
// Bp[((kt*NT + nt)*32 + lane)*8 + j] = pack(W[kt*32+16*kg+2j][n],
//                                           W[kt*32+16*kg+2j+1][n])
// kg = lane>>4, n = nt*16 + (lane&15).
__global__ void k_pack_w(const float* __restrict__ W,
                         unsigned int* __restrict__ Bp, int K, int NF) {
    int idx = blockIdx.x * blockDim.x + threadIdx.x;
    int total = (K / 32) * (NF / 16) * 32 * 8;
    if (idx >= total) return;
    int j    =  idx       & 7;
    int lane = (idx >> 3) & 31;
    int t    =  idx >> 8;                 // kt*(NF/16) + nt
    int nt   =  t % (NF / 16);
    int kt   =  t / (NF / 16);
    int kg   =  lane >> 4;
    int n    =  nt * 16 + (lane & 15);
    int k    =  kt * 32 + 16 * kg + 2 * j;
    unsigned int lo = f32_to_bf16_bits(W[k * NF + n]);
    unsigned int hi = f32_to_bf16_bits(W[(k + 1) * NF + n]);
    Bp[idx] = lo | (hi << 16);
}

// ---- WMMA GEMM:  C[N,64] = A[N,K](bf16) @ W[K,64] --------------------------
// Block = 128 threads (4 waves). 16-row strip of A pulled into LDS with
// async-to-LDS B128 transfers; wave w computes the 16x16 tile for columns
// [16w, 16w+16).
template <int K>
__global__ __launch_bounds__(128) void k_gemm_wmma(
        const unsigned short* __restrict__ A,   // bf16 [N,K]
        const unsigned int*   __restrict__ Bp,  // packed B fragments
        float* __restrict__ C) {                // f32 [N,64]
    constexpr int KW = K / 2;                   // 32-bit words per A row
    constexpr int CH = KW / 4;                  // 16B chunks per A row
    // +4 word pad keeps every row base 16B-aligned for async B128 writes.
    __shared__ unsigned int sA[16][KW + 4];

    const int tid  = threadIdx.x;
    const int wave = tid >> 5;                  // column tile (0..3)
    const int lane = tid & 31;
    const int row0 = blockIdx.x * 16;

    // ---- async strip load: 16 rows x CH 16-byte chunks -> LDS -------------
    const char* Abase = (const char*)(A + (size_t)row0 * K);
    for (int i = tid; i < 16 * CH; i += 128) {
        int r = i / CH, c = i % CH;
        unsigned long long ga =
            (unsigned long long)(Abase + (size_t)r * (K * 2) + c * 16);
        unsigned lo = (unsigned)(unsigned long long)&sA[r][c * 4];
        asm volatile("global_load_async_to_lds_b128 %0, %1, off"
                     :: "v"(lo), "v"(ga) : "memory");
    }
    asm volatile("s_wait_asynccnt 0x0" ::: "memory");
    __syncthreads();

    const int m  = lane & 15;                   // A row within strip
    const int kg = lane >> 4;                   // K group select
    const int n  = lane & 15;                   // output column within tile

    v8f acc = {};
    #pragma unroll
    for (int kt = 0; kt < K / 32; ++kt) {
        ABu a, b;
        // A fragment: word offsets kt*16 + 8*kg + {0..7}
        int base = kt * 16 + 8 * kg;
        #pragma unroll
        for (int q = 0; q < 8; ++q) {
            int wi = base + q;
            a.u[q] = sA[m][wi];
        }
        // B fragment: one contiguous 32B load per lane from packed buffer
        const unsigned int* bsrc =
            Bp + (((size_t)kt * 4 + wave) * 32 + lane) * 8;
        b.u = *(const v8u*)bsrc;

        acc = __builtin_amdgcn_wmma_f32_16x16x32_bf16(
                  false, a.h, false, b.h, (short)0, acc, false, false);
    }

    // C/D layout: VGPR r -> M = r + 8*kg, N = lane&15
    float* Crow = C + (size_t)row0 * 64 + wave * 16 + n;
    #pragma unroll
    for (int r = 0; r < 8; ++r)
        Crow[(size_t)(r + 8 * kg) * 64] = acc[r];
}

// ---- zero accumulator ------------------------------------------------------
__global__ void k_zero(float* __restrict__ H, int n) {
    int i = blockIdx.x * 256 + threadIdx.x;
    if (i < n) H[i] = 0.0f;
}

// ---- edge scatter: H[dst] += w * P[src]  (wave per edge, 2 feats/lane) -----
__global__ void k_scatter(const long long* __restrict__ ei,   // [2,E] i64
                          const float* __restrict__ ew,
                          const float* __restrict__ P,
                          float* __restrict__ H, int nE) {
    int e    = blockIdx.x * 8 + (threadIdx.x >> 5);
    int lane = threadIdx.x & 31;
    if (e >= nE) return;
    __builtin_prefetch(&ei[e + 2048], 0, 0);    // global_prefetch_b8
    int   s = (int)ei[e];
    int   d = (int)ei[(size_t)nE + e];
    float w = ew[e];
    const float* p = P + (size_t)s * 64;
    float m0 = p[lane] * w;
    float m1 = p[lane + 32] * w;
    atomicAdd(&H[(size_t)d * 64 + lane],      m0);
    atomicAdd(&H[(size_t)d * 64 + lane + 32], m1);
}

// ---- fused bias(+residual) + layernorm + relu -> bf16 ----------------------
// mode 0:  v = agg + b1 ; ORI = v           (first propagation step)
// mode 1:  v = agg + b2 + 0.5*ORI           (later steps)
// then Z = bf16(relu(layernorm(v)));  H itself is NOT materialized here
// (the next scatter re-zeroes it), saving a full 51MB sweep per step.
__global__ void k_ln_relu_fused(const float* __restrict__ Hagg,
                                const float* __restrict__ bias,
                                float* __restrict__ ORI,
                                const float* __restrict__ g,
                                const float* __restrict__ be,
                                unsigned short* __restrict__ Z,
                                int nRows, int mode) {
    int row  = blockIdx.x * 8 + (threadIdx.x >> 5);
    int lane = threadIdx.x & 31;
    if (row >= nRows) return;
    size_t o0 = (size_t)row * 64 + lane;
    size_t o1 = o0 + 32;
    float v0 = Hagg[o0] + bias[lane];
    float v1 = Hagg[o1] + bias[lane + 32];
    if (mode == 0) { ORI[o0] = v0; ORI[o1] = v1; }
    else           { v0 += 0.5f * ORI[o0]; v1 += 0.5f * ORI[o1]; }
    float s  = v0 + v1;
    float ss = v0 * v0 + v1 * v1;
    #pragma unroll
    for (int off = 16; off; off >>= 1) {        // wave32 reduction
        s  += __shfl_xor(s,  off);
        ss += __shfl_xor(ss, off);
    }
    float mu   = s * (1.0f / 64.0f);
    float var  = ss * (1.0f / 64.0f) - mu * mu;
    float rstd = rsqrtf(var + 1e-5f);
    float z0 = fmaxf((v0 - mu) * rstd * g[lane]      + be[lane],      0.0f);
    float z1 = fmaxf((v1 - mu) * rstd * g[lane + 32] + be[lane + 32], 0.0f);
    Z[o0] = f32_to_bf16_bits(z0);
    Z[o1] = f32_to_bf16_bits(z1);
}

// ---- final:  H = agg + b2 + 0.5*ORI ----------------------------------------
__global__ void k_final(float* __restrict__ H, const float* __restrict__ bias,
                        const float* __restrict__ ORI, int n) {
    int i = blockIdx.x * 256 + threadIdx.x;
    if (i < n) H[i] = H[i] + bias[i & 63] + 0.5f * ORI[i];
}

// ---------------------------------------------------------------------------
extern "C" void kernel_launch(void* const* d_in, const int* in_sizes, int n_in,
                              void* d_out, int out_size, void* d_ws, size_t ws_size,
                              hipStream_t stream) {
    const float*     x  = (const float*)d_in[0];
    const long long* ei = (const long long*)d_in[1];
    const float*     ew = (const float*)d_in[2];
    const float*     W1 = (const float*)d_in[3];
    const float*     b1 = (const float*)d_in[4];
    const float*     W2 = (const float*)d_in[5];
    const float*     b2 = (const float*)d_in[6];
    const float*     lg = (const float*)d_in[7];
    const float*     lb = (const float*)d_in[8];

    const int IN = 128, HF = 64;
    const int N  = in_sizes[0] / IN;            // 100000 (divisible by 16)
    const int E  = in_sizes[1] / 2;             // 1200000

    // workspace carve-out (~90 MB total)
    char* ws = (char*)d_ws;
    auto carve = [&](size_t bytes) {
        char* p = ws; ws += (bytes + 255) & ~(size_t)255; return p;
    };
    unsigned short* Xb  = (unsigned short*)carve((size_t)N * IN * 2);
    float*          P   = (float*)         carve((size_t)N * HF * 4);
    float*          ORI = (float*)         carve((size_t)N * HF * 4);
    unsigned short* Zb  = (unsigned short*)carve((size_t)N * HF * 2);
    unsigned int*   W1p = (unsigned int*)  carve((size_t)IN * HF * 2);
    unsigned int*   W2p = (unsigned int*)  carve((size_t)HF * HF * 2);
    float* H = (float*)d_out;

    const int nElem = N * HF;

    // precision staging
    k_cvt_bf16<<<(N * IN + 255) / 256, 256, 0, stream>>>(x, Xb, N * IN);
    k_pack_w<<<(4 * 4 * 32 * 8 + 255) / 256, 256, 0, stream>>>(W1, W1p, IN, HF);
    k_pack_w<<<(2 * 4 * 32 * 8 + 255) / 256, 256, 0, stream>>>(W2, W2p, HF, HF);

    // conv1: agg1 = segment_sum(w * (x@W1)[src])
    k_gemm_wmma<128><<<N / 16, 128, 0, stream>>>(Xb, W1p, P);
    k_zero<<<(nElem + 255) / 256, 256, 0, stream>>>(H, nElem);
    k_scatter<<<(E + 7) / 8, 256, 0, stream>>>(ei, ew, P, H, E);

    // step 2:  h1 = agg1+b1 (saved as ORI); Z = relu(LN(h1)); agg2
    k_ln_relu_fused<<<(N + 7) / 8, 256, 0, stream>>>(H, b1, ORI, lg, lb, Zb, N, 0);
    k_gemm_wmma<64><<<N / 16, 128, 0, stream>>>(Zb, W2p, P);
    k_zero<<<(nElem + 255) / 256, 256, 0, stream>>>(H, nElem);
    k_scatter<<<(E + 7) / 8, 256, 0, stream>>>(ei, ew, P, H, E);

    // step 3:  h2 = agg2+b2+0.5*ORI; Z = relu(LN(h2)); agg3
    k_ln_relu_fused<<<(N + 7) / 8, 256, 0, stream>>>(H, b2, ORI, lg, lb, Zb, N, 1);
    k_gemm_wmma<64><<<N / 16, 128, 0, stream>>>(Zb, W2p, P);
    k_zero<<<(nElem + 255) / 256, 256, 0, stream>>>(H, nElem);
    k_scatter<<<(E + 7) / 8, 256, 0, stream>>>(ei, ew, P, H, E);

    // output: h3 = agg3 + b2 + 0.5*ORI
    k_final<<<(nElem + 255) / 256, 256, 0, stream>>>(H, b2, ORI, nElem);
}
